// GCNN_29738353558039
// MI455X (gfx1250) — matrix operand order repaired and checked by reading, not compiled, not run on hardware
//
#include <hip/hip_runtime.h>

typedef __attribute__((ext_vector_type(16))) _Float16 v16h;
typedef __attribute__((ext_vector_type(8)))  _Float16 v8h;
typedef __attribute__((ext_vector_type(8)))  float    v8f;

#define DOUT 256  // all layer outputs are 256 wide

// ---------------- elementwise / setup kernels ----------------

__global__ void k_f32_to_f16(const float* __restrict__ src, _Float16* __restrict__ dst, size_t n) {
  size_t i = (size_t)blockIdx.x * blockDim.x + threadIdx.x;
  if (i < n) dst[i] = (_Float16)src[i];
}

// W [K x N] f32 (row-major) -> WT [N x K] f16 (row-major)
__global__ void k_w_to_f16T(const float* __restrict__ W, _Float16* __restrict__ WT, int K, int N) {
  int i = blockIdx.x * blockDim.x + threadIdx.x;
  if (i < K * N) {
    int k = i / N, n = i % N;
    WT[(size_t)n * K + k] = (_Float16)W[i];
  }
}

__global__ void k_fill1(float* __restrict__ p, int n) {
  int i = blockIdx.x * blockDim.x + threadIdx.x;
  if (i < n) p[i] = 1.0f;  // self-loop contribution to degree
}

__global__ void k_zero_i32(int* __restrict__ p, int n) {
  int i = blockIdx.x * blockDim.x + threadIdx.x;
  if (i < n) p[i] = 0;
}

__global__ void k_cvt_idx(const long long* __restrict__ ei, int* __restrict__ rowi,
                          int* __restrict__ coli, int E) {
  int e = blockIdx.x * blockDim.x + threadIdx.x;
  if (e < E) {
    rowi[e] = (int)ei[e];
    coli[e] = (int)ei[(size_t)E + e];
  }
}

__global__ void k_deg_edges(const int* __restrict__ coli, const float* __restrict__ ew,
                            float* __restrict__ deg, int E) {
  int e = blockIdx.x * blockDim.x + threadIdx.x;
  if (e < E) atomicAdd(deg + coli[e], ew[e]);
}

__global__ void k_rsqrt_inplace(float* __restrict__ p, int n) {
  int i = blockIdx.x * blockDim.x + threadIdx.x;
  if (i < n) p[i] = rsqrtf(p[i]);  // deg >= 1 always (self loop)
}

// ---------------- CSR build (by destination column) ----------------

__global__ void k_count(const int* __restrict__ coli, int* __restrict__ cnt, int E) {
  int e = blockIdx.x * blockDim.x + threadIdx.x;
  if (e < E) atomicAdd(cnt + coli[e], 1);
}

// block-level exclusive scan of 256 elems; writes per-block total to part[]
__global__ void __launch_bounds__(256)
k_scan_block(const int* __restrict__ in, int* __restrict__ out, int* __restrict__ part, int n) {
  __shared__ int s[256];
  int tid = threadIdx.x;
  int g = blockIdx.x * 256 + tid;
  int v = (g < n) ? in[g] : 0;
  s[tid] = v;
  __syncthreads();
#pragma unroll
  for (int d = 1; d < 256; d <<= 1) {
    int t = (tid >= d) ? s[tid - d] : 0;
    __syncthreads();
    s[tid] += t;
    __syncthreads();
  }
  if (g < n) out[g] = s[tid] - v;           // exclusive
  if (tid == 255) part[blockIdx.x] = s[255]; // block total (inclusive of last)
}

// single-block exclusive scan of up to 512 partials
__global__ void __launch_bounds__(512)
k_scan_part(int* __restrict__ part, int nb) {
  __shared__ int s[512];
  int tid = threadIdx.x;
  int v = (tid < nb) ? part[tid] : 0;
  s[tid] = v;
  __syncthreads();
#pragma unroll
  for (int d = 1; d < 512; d <<= 1) {
    int t = (tid >= d) ? s[tid - d] : 0;
    __syncthreads();
    s[tid] += t;
    __syncthreads();
  }
  if (tid < nb) part[tid] = s[tid] - v;  // exclusive
}

// finalize rowptr = local_excl + block_offset, and copy to cursor
__global__ void k_scan_add(int* __restrict__ rowptr, int* __restrict__ cur,
                           const int* __restrict__ part, int n) {
  int g = blockIdx.x * blockDim.x + threadIdx.x;
  if (g < n) {
    int v = rowptr[g] + part[g >> 8];
    rowptr[g] = v;
    cur[g] = v;
  }
}

// bucket edges by destination; store source row + fully normalized edge weight
__global__ void k_fill_csr(const int* __restrict__ rowi, const int* __restrict__ coli,
                           const float* __restrict__ ew, const float* __restrict__ dinv,
                           int* __restrict__ cur, int* __restrict__ esrc,
                           float* __restrict__ eval, int E) {
  int e = blockIdx.x * blockDim.x + threadIdx.x;
  if (e < E) {
    int r = rowi[e], c = coli[e];
    int pos = atomicAdd(cur + c, 1);
    esrc[pos] = r;
    eval[pos] = dinv[r] * ew[e] * dinv[c];
  }
}

// ---------------- fused aggregation: gather-by-destination (no atomics) ----------------
// one wave per node: out[c,:] = sum_j eval[j]*h[esrc[j],:] + dinv[c]^2*h[c,:] + b
// then optional ReLU; writes f32 (out32) and/or f16 (out16).
__global__ void __launch_bounds__(256)
k_agg_csr(const int* __restrict__ rowptr, const int* __restrict__ esrc,
          const float* __restrict__ eval, const float* __restrict__ h,
          const float* __restrict__ dinv, const float* __restrict__ bias,
          float* __restrict__ out32, _Float16* __restrict__ out16,
          int Nn, int E, int relu) {
  int node = blockIdx.x * (blockDim.x >> 5) + (threadIdx.x >> 5);
  int lane = threadIdx.x & 31;
  if (node >= Nn) return;

  int beg = rowptr[node];
  int end = (node == Nn - 1) ? E : rowptr[node + 1];
  float d = dinv[node];
  float dd = d * d;

  const float4* hp = (const float4*)(h + (size_t)node * DOUT);
  float4 s0 = hp[lane * 2 + 0];
  float4 s1 = hp[lane * 2 + 1];
  const float4* bp = (const float4*)(bias + lane * 8);
  float4 b0 = bp[0], b1 = bp[1];

  float a0 = dd * s0.x + b0.x, a1 = dd * s0.y + b0.y;
  float a2 = dd * s0.z + b0.z, a3 = dd * s0.w + b0.w;
  float a4 = dd * s1.x + b1.x, a5 = dd * s1.y + b1.y;
  float a6 = dd * s1.z + b1.z, a7 = dd * s1.w + b1.w;

  int j = beg;
  for (; j + 1 < end; j += 2) {  // 2-edge unroll to overlap gather latency
    int   r0 = esrc[j],     r1 = esrc[j + 1];
    float w0 = eval[j],     w1 = eval[j + 1];
    const float4* p0 = (const float4*)(h + (size_t)r0 * DOUT);
    const float4* p1 = (const float4*)(h + (size_t)r1 * DOUT);
    float4 u0 = p0[lane * 2 + 0], u1 = p0[lane * 2 + 1];
    float4 q0 = p1[lane * 2 + 0], q1 = p1[lane * 2 + 1];
    a0 += w0 * u0.x + w1 * q0.x;  a1 += w0 * u0.y + w1 * q0.y;
    a2 += w0 * u0.z + w1 * q0.z;  a3 += w0 * u0.w + w1 * q0.w;
    a4 += w0 * u1.x + w1 * q1.x;  a5 += w0 * u1.y + w1 * q1.y;
    a6 += w0 * u1.z + w1 * q1.z;  a7 += w0 * u1.w + w1 * q1.w;
  }
  if (j < end) {
    int r0 = esrc[j];
    float w0 = eval[j];
    const float4* p0 = (const float4*)(h + (size_t)r0 * DOUT);
    float4 u0 = p0[lane * 2 + 0], u1 = p0[lane * 2 + 1];
    a0 += w0 * u0.x;  a1 += w0 * u0.y;  a2 += w0 * u0.z;  a3 += w0 * u0.w;
    a4 += w0 * u1.x;  a5 += w0 * u1.y;  a6 += w0 * u1.z;  a7 += w0 * u1.w;
  }

  if (relu) {
    a0 = fmaxf(a0, 0.f); a1 = fmaxf(a1, 0.f); a2 = fmaxf(a2, 0.f); a3 = fmaxf(a3, 0.f);
    a4 = fmaxf(a4, 0.f); a5 = fmaxf(a5, 0.f); a6 = fmaxf(a6, 0.f); a7 = fmaxf(a7, 0.f);
  }
  size_t base = (size_t)node * DOUT + lane * 8;
  if (out32) {
    float4* o = (float4*)(out32 + base);
    o[0] = make_float4(a0, a1, a2, a3);
    o[1] = make_float4(a4, a5, a6, a7);
  }
  if (out16) {
    v8h o16;
    o16[0] = (_Float16)a0; o16[1] = (_Float16)a1; o16[2] = (_Float16)a2; o16[3] = (_Float16)a3;
    o16[4] = (_Float16)a4; o16[5] = (_Float16)a5; o16[6] = (_Float16)a6; o16[7] = (_Float16)a7;
    *(v8h*)(out16 + base) = o16;
  }
}

// ---------------- WMMA GEMM: C[M x 256] = A[M x K] * B[K x 256] ----------------
// A: f16 row-major.  BT: B transposed, f16 [256 x K] row-major.  C: f32.
// wave = 32 rows x 128 cols (2 m-tiles x 8 n-tiles); block = 4 waves = 128 rows;
// grid.y = 2 n-halves.  B fragments software-pipelined (prefetch t+1), each B
// fragment feeds 2 WMMAs.
__device__ __forceinline__ v16h load_frag(const _Float16* p, int off2) {
  v8h x0 = *(const v8h*)(p);
  v8h x1 = *(const v8h*)(p + off2);
  v16h f;
#pragma unroll
  for (int i = 0; i < 8; ++i) { f[i] = x0[i]; f[i + 8] = x1[i]; }
  return f;
}

__global__ void __launch_bounds__(128)
k_gemm_wmma(const _Float16* __restrict__ A, const _Float16* __restrict__ BT,
            float* __restrict__ C, int M, int K) {
  const int lane = threadIdx.x & 31;
  const int wave = threadIdx.x >> 5;
  const int l  = lane & 15;
  const int hi = lane >> 4;
  const int mbase = (blockIdx.x * 4 + wave) * 32;
  const int nbase = blockIdx.y * 128;
  if (mbase >= M) return;  // wave-uniform exit; surviving waves keep EXEC all-1s

  int ar0 = mbase + l;       if (ar0 > M - 1) ar0 = M - 1;
  int ar1 = mbase + 16 + l;  if (ar1 > M - 1) ar1 = M - 1;
  // A frag: halves 0-7 = K hi*8+0..7, halves 8-15 = K 16+hi*8+0..7
  const _Float16* ap0 = A + (size_t)ar0 * K + hi * 8;
  const _Float16* ap1 = A + (size_t)ar1 * K + hi * 8;
  // B frag: halves 0-7 = K hi*8+0..7, halves 8-15 = K 16+hi*8+0..7
  const _Float16* bp = BT + (size_t)(nbase + l) * K + hi * 8;
  const size_t btile = (size_t)16 * K;  // pointer stride per n-tile

  v8f zero = {};
  v8f acc0[8], acc1[8];
#pragma unroll
  for (int t = 0; t < 8; ++t) { acc0[t] = zero; acc1[t] = zero; }

  for (int k0 = 0; k0 < K; k0 += 32) {
    v16h a0f = load_frag(ap0 + k0, 16);
    v16h a1f = load_frag(ap1 + k0, 16);
    v16h bcur = load_frag(bp + k0, 16);
#pragma unroll
    for (int t = 0; t < 8; ++t) {
      v16h bnext = bcur;
      if (t < 7) bnext = load_frag(bp + (size_t)(t + 1) * btile + k0, 16);  // prefetch
      acc0[t] = __builtin_amdgcn_wmma_f32_16x16x32_f16(
          false, a0f, false, bcur, (short)0, acc0[t], false, false);
      acc1[t] = __builtin_amdgcn_wmma_f32_16x16x32_f16(
          false, a1f, false, bcur, (short)0, acc1[t], false, false);
      bcur = bnext;
    }
  }

  // C frag: VGPR i -> row tilebase + i + hi*8, col = l within tile
#pragma unroll
  for (int t = 0; t < 8; ++t) {
    int ncol = nbase + t * 16 + l;
#pragma unroll
    for (int i = 0; i < 8; ++i) {
      int r0 = mbase + i + hi * 8;
      int r1 = mbase + 16 + i + hi * 8;
      if (r0 < M) C[(size_t)r0 * DOUT + ncol] = acc0[t][i];
      if (r1 < M) C[(size_t)r1 * DOUT + ncol] = acc1[t][i];
    }
  }
}

// ---------------- driver ----------------

static inline size_t align256(size_t x) { return (x + 255) & ~(size_t)255; }

extern "C" void kernel_launch(void* const* d_in, const int* in_sizes, int n_in,
                              void* d_out, int out_size, void* d_ws, size_t ws_size,
                              hipStream_t stream) {
  const float*     x  = (const float*)d_in[0];
  const long long* ei = (const long long*)d_in[1];  // int64 edge_index [2, E]
  const float*     ew = (const float*)d_in[2];
  const float*     W1 = (const float*)d_in[3];
  const float*     b1 = (const float*)d_in[4];
  const float*     W2 = (const float*)d_in[5];
  const float*     b2 = (const float*)d_in[6];
  const float*     W3 = (const float*)d_in[7];
  const float*     b3 = (const float*)d_in[8];

  const int N = in_sizes[0] / 512;  // 100000
  const int E = in_sizes[2];        // 1600000
  const int DIN = 512;

  // workspace carve-up
  char* ws = (char*)d_ws;
  size_t off = 0;
  auto carve = [&](size_t bytes) -> void* {
    void* p = ws + off;
    off = align256(off + bytes);
    return p;
  };
  _Float16* X16    = (_Float16*)carve((size_t)N * DIN * 2);   // 102.4 MB
  float*    h32    = (float*)   carve((size_t)N * DOUT * 4);  // 102.4 MB
  _Float16* act16  = (_Float16*)carve((size_t)N * DOUT * 2);  //  51.2 MB
  _Float16* W1T    = (_Float16*)carve((size_t)DOUT * DIN * 2);
  _Float16* W2T    = (_Float16*)carve((size_t)DOUT * DOUT * 2);
  _Float16* W3T    = (_Float16*)carve((size_t)DOUT * DOUT * 2);
  float*    dinv   = (float*)   carve((size_t)N * 4);
  int*      rowi   = (int*)     carve((size_t)E * 4);
  int*      coli   = (int*)     carve((size_t)E * 4);
  int*      cnt    = (int*)     carve((size_t)N * 4);
  int*      rowptr = (int*)     carve((size_t)N * 4);
  int*      cur    = (int*)     carve((size_t)N * 4);
  int*      part   = (int*)     carve((size_t)512 * 4);
  int*      esrc   = (int*)     carve((size_t)E * 4);
  float*    evalw  = (float*)   carve((size_t)E * 4);
  (void)ws_size; (void)n_in;

  const int T = 256;
  size_t nx = (size_t)N * DIN;
  size_t nh = (size_t)N * DOUT;
  int NB = (N + 255) / 256;  // scan blocks (391 <= 512)

  // ---- precompute: f16 conversions, degree norms, CSR by destination ----
  k_f32_to_f16<<<(unsigned)((nx + T - 1) / T), T, 0, stream>>>(x, X16, nx);
  k_w_to_f16T<<<(DIN * DOUT + T - 1) / T, T, 0, stream>>>(W1, W1T, DIN, DOUT);
  k_w_to_f16T<<<(DOUT * DOUT + T - 1) / T, T, 0, stream>>>(W2, W2T, DOUT, DOUT);
  k_w_to_f16T<<<(DOUT * DOUT + T - 1) / T, T, 0, stream>>>(W3, W3T, DOUT, DOUT);
  k_cvt_idx<<<(E + T - 1) / T, T, 0, stream>>>(ei, rowi, coli, E);
  k_fill1<<<(N + T - 1) / T, T, 0, stream>>>(dinv, N);
  k_deg_edges<<<(E + T - 1) / T, T, 0, stream>>>(coli, ew, dinv, E);
  k_rsqrt_inplace<<<(N + T - 1) / T, T, 0, stream>>>(dinv, N);
  k_zero_i32<<<(N + T - 1) / T, T, 0, stream>>>(cnt, N);
  k_count<<<(E + T - 1) / T, T, 0, stream>>>(coli, cnt, E);
  k_scan_block<<<NB, 256, 0, stream>>>(cnt, rowptr, part, N);
  k_scan_part<<<1, 512, 0, stream>>>(part, NB);
  k_scan_add<<<(N + T - 1) / T, T, 0, stream>>>(rowptr, cur, part, N);
  k_fill_csr<<<(E + T - 1) / T, T, 0, stream>>>(rowi, coli, ew, dinv, cur, esrc, evalw, E);

  dim3 gemmGrid((N + 127) / 128, 2);
  unsigned aggBlocks = (unsigned)((N + 7) / 8);  // 8 waves/block, 1 wave/node

  // ---- layer 1 ----
  k_gemm_wmma<<<gemmGrid, 128, 0, stream>>>(X16, W1T, h32, N, DIN);
  k_agg_csr<<<aggBlocks, T, 0, stream>>>(rowptr, esrc, evalw, h32, dinv, b1,
                                         nullptr, act16, N, E, 1);
  // ---- layer 2 ----
  k_gemm_wmma<<<gemmGrid, 128, 0, stream>>>(act16, W2T, h32, N, DOUT);
  k_agg_csr<<<aggBlocks, T, 0, stream>>>(rowptr, esrc, evalw, h32, dinv, b2,
                                         nullptr, act16, N, E, 1);
  // ---- layer 3 (no relu) -> d_out (f32) ----
  k_gemm_wmma<<<gemmGrid, 128, 0, stream>>>(act16, W3T, h32, N, DOUT);
  k_agg_csr<<<aggBlocks, T, 0, stream>>>(rowptr, esrc, evalw, h32, dinv, b3,
                                         (float*)d_out, nullptr, N, E, 0);
}
